// GNNEncoder_3315714752917
// MI455X (gfx1250) — matrix operand order repaired and checked by reading, not compiled, not run on hardware
//
#include <hip/hip_runtime.h>
#include <hip/hip_bf16.h>
#include <math.h>

typedef __attribute__((ext_vector_type(16))) _Float16 v16h;
typedef __attribute__((ext_vector_type(8)))  float    v8f;

#define S 32
#define NFRAG 44        // 6 msg + 36 gru + 2 head pre-packed B fragments
#define LDS_PITCH 36    // floats per LDS row: 16B-aligned rows, conflict-friendly
#define EPW 8           // edges per wave in scatter kernel

__device__ __forceinline__ float sigmoidf(float x) {
    return 1.0f / (1.0f + __expf(-x));
}

__device__ __forceinline__ v8f splat8(float x) {
    v8f c;
#pragma unroll
    for (int v = 0; v < 8; ++v) c[v] = x;
    return c;
}

// ---- WMMA fragment helpers (A/B use contiguous k = 16*(lane/16) + elem) -------------------------

// A fragment: 16 rows x K=32 from f32 row-major (ld=32), converted to f16.
//   lanes 0-15 : row=lane,    K 0..15 -> elems 0-15
//   lanes 16-31: row=lane-16, K 16..31 -> elems 0-15
__device__ __forceinline__ v16h load_a_f16(const float* base, int row_base, int nrows) {
    const int lane = threadIdx.x & 31;
    int m = row_base + (lane & 15);
    if (m >= nrows) m = nrows - 1;          // clamp; stores masked on tail
    const float* p = base + (size_t)m * S;
    const int kb = (lane & 16) ? 16 : 0;
    v16h a;
#pragma unroll
    for (int e = 0; e < 8; ++e) a[e]     = (_Float16)p[kb + e];
#pragma unroll
    for (int e = 0; e < 8; ++e) a[e + 8] = (_Float16)p[kb + 8 + e];
    return a;
}

// Pre-packed B fragment: contiguous 32B per lane, coalesced across wave.
__device__ __forceinline__ v16h load_packed(const _Float16* __restrict__ packed, int frag) {
    const int lane = threadIdx.x & 31;
    return *(const v16h*)(packed + ((size_t)frag * 32 + lane) * 16);
}

__device__ __forceinline__ v8f wmma16(v16h a, v16h b, v8f c) {
    return __builtin_amdgcn_wmma_f32_16x16x32_f16(false, a, false, b, (short)0, c,
                                                  false, false);
}

// Masked (tail) store: VGPR v -> row v (+8 for upper lanes), lane%16 -> col.
__device__ __forceinline__ void store_d_masked(float* out, int ld, int row_base, int nrows,
                                               int jbase, v8f c, bool do_relu) {
    const int lane = threadIdx.x & 31;
    const int j    = jbase + (lane & 15);
    const int mofs = (lane & 16) ? 8 : 0;
#pragma unroll
    for (int v = 0; v < 8; ++v) {
        int m = row_base + mofs + v;
        float val = c[v];
        if (do_relu) val = fmaxf(val, 0.0f);
        if (m < nrows) out[(size_t)m * ld + j] = val;
    }
}

// Full-tile strided store (no masking, straight-line).
__device__ __forceinline__ void store_d_full(float* out, int ld, int row_base,
                                             int jbase, v8f c, bool do_relu) {
    const int lane = threadIdx.x & 31;
    const int mofs = (lane & 16) ? 8 : 0;
    float* p = out + (size_t)(row_base + mofs) * ld + jbase + (lane & 15);
#pragma unroll
    for (int v = 0; v < 8; ++v) {
        float val = c[v];
        if (do_relu) val = fmaxf(val, 0.0f);
        p[(size_t)v * ld] = val;
    }
}

// Full 16x32 tile store, transposed through LDS -> 4x global_store_b128/lane.
// ldsw: per-wave scratch, 16 rows x LDS_PITCH floats, 16B-aligned.
__device__ __forceinline__ void store_tile32_coalesced(float* out, int row_base,
                                                       v8f c0, v8f c1, bool do_relu,
                                                       float* ldsw) {
    const int lane = threadIdx.x & 31;
    const int jl   = lane & 15;
    const int mofs = (lane & 16) ? 8 : 0;
#pragma unroll
    for (int v = 0; v < 8; ++v) {
        float a0 = c0[v], a1 = c1[v];
        if (do_relu) { a0 = fmaxf(a0, 0.0f); a1 = fmaxf(a1, 0.0f); }
        ldsw[(mofs + v) * LDS_PITCH + jl]      = a0;   // cols 0-15
        ldsw[(mofs + v) * LDS_PITCH + 16 + jl] = a1;   // cols 16-31
    }
    // In-wave LDS RAW: DS ops are in-order per wave; wait for store data, and
    // fence the compiler so the loads below aren't hoisted.
    asm volatile("s_wait_dscnt 0x0" ::: "memory");
    const int half = lane >> 4;                        // 0: cols 0-15, 1: cols 16-31
    const float* src = ldsw + jl * LDS_PITCH + half * 16;
    float* gout = out + (size_t)(row_base + jl) * S + half * 16;
#pragma unroll
    for (int q = 0; q < 4; ++q) {
        float4 t = *(const float4*)(src + q * 4);
        *(float4*)(gout + q * 4) = t;                  // b128, coalesced across wave
    }
}

// ---------------- kernels ----------------

// One-shot: swizzle all B operands into WMMA lane layout as f16.
// frag 0..5   : msg_W[r] (K-major, ld=32), r=f/2, col tile t=f%2
// frag 6..41  : GRU, idx=f-6, r=idx/12, w=idx%12, which=w/6 (0=wih,1=whh),
//               col tile gc=w%6 over the 96 output cols; B[k][q]=W[q*32+k]
// frag 42,43  : mu_W / ls_W (K-major, ld=16)
__global__ __launch_bounds__(256) void pack_b_kernel(const float* __restrict__ msg_W,
                                                     const float* __restrict__ g_wih,
                                                     const float* __restrict__ g_whh,
                                                     const float* __restrict__ muW,
                                                     const float* __restrict__ lsW,
                                                     _Float16* __restrict__ packed) {
    int f = blockIdx.x * (blockDim.x >> 5) + (threadIdx.x >> 5);
    if (f >= NFRAG) return;
    const int lane = threadIdx.x & 31;
    const int jl = lane & 15;
    const int kb = (lane & 16) ? 16 : 0;
    v16h b;
    if (f < 6) {
        int r = f >> 1, t = f & 1;
        const float* W = msg_W + (size_t)r * S * S;
        int j = t * 16 + jl;
#pragma unroll
        for (int e = 0; e < 16; ++e) b[e] = (_Float16)W[(size_t)(kb + e) * S + j];
    } else if (f < 42) {
        int idx = f - 6, r = idx / 12, w = idx % 12, which = w / 6, gc = w % 6;
        const float* Wt = (which ? g_whh : g_wih) + (size_t)r * 3 * S * S;
        const float* p = Wt + (size_t)(gc * 16 + jl) * S + kb;  // 16 contiguous
#pragma unroll
        for (int e = 0; e < 16; ++e) b[e] = (_Float16)p[e];
    } else {
        const float* W = (f == 42) ? muW : lsW;
#pragma unroll
        for (int e = 0; e < 16; ++e) b[e] = (_Float16)W[(size_t)(kb + e) * 16 + jl];
    }
    *(v16h*)(packed + ((size_t)f * 32 + lane) * 16) = b;
}

// state = relu(x @ W_in + b_in), x:[N,7]
__global__ __launch_bounds__(256) void input_net_kernel(const float* __restrict__ x,
                                                        const float* __restrict__ W_in,
                                                        const float* __restrict__ b_in,
                                                        float* __restrict__ state, int N) {
    size_t tid = (size_t)blockIdx.x * blockDim.x + threadIdx.x;
    if (tid >= (size_t)N * S) return;
    int n = (int)(tid >> 5);
    int s = (int)(tid & 31);
    const float* xr = x + (size_t)n * 7;
    float acc = b_in[s];
#pragma unroll
    for (int f = 0; f < 7; ++f) acc = fmaf(xr[f], W_in[f * S + s], acc);
    state[tid] = fmaxf(acc, 0.0f);
}

// msg = relu(state @ W + b); also zeroes this tile's rows of agg (fused with
// the WMMA latency; removes a separate full pass over agg each round).
__global__ __launch_bounds__(256) void message_kernel(const float* __restrict__ state,
                                                      const _Float16* __restrict__ packed,
                                                      int fb,
                                                      const float* __restrict__ b,
                                                      float* __restrict__ msg,
                                                      float* __restrict__ agg, int N) {
    __shared__ __align__(16) float lds[8][16 * LDS_PITCH];
    int wave = blockIdx.x * (blockDim.x >> 5) + (threadIdx.x >> 5);
    int row_base = wave * 16;
    if (row_base >= N) return;                 // wave-uniform; EXEC full at WMMA
    const int lane = threadIdx.x & 31;
    const int jl = lane & 15;
    const bool full = (row_base + 16 <= N);

    v16h a = load_a_f16(state, row_base, N);
    v8f c0 = wmma16(a, load_packed(packed, fb + 0), splat8(b[jl]));
    v8f c1 = wmma16(a, load_packed(packed, fb + 1), splat8(b[16 + jl]));

    if (full) {
        // zero agg tile: lane -> row jl, half (lane>>4): 4x b128 zero stores
        const float4 z = make_float4(0.0f, 0.0f, 0.0f, 0.0f);
        float* az = agg + (size_t)(row_base + jl) * S + (lane >> 4) * 16;
#pragma unroll
        for (int q = 0; q < 4; ++q) *(float4*)(az + q * 4) = z;
        store_tile32_coalesced(msg, row_base, c0, c1, true, lds[threadIdx.x >> 5]);
    } else {
#pragma unroll
        for (int v = 0; v < 8; ++v) {
            int m = row_base + ((lane & 16) ? 8 : 0) + v;
            if (m < N) {
                agg[(size_t)m * S + jl] = 0.0f;
                agg[(size_t)m * S + 16 + jl] = 0.0f;
            }
        }
        store_d_masked(msg, S, row_base, N, 0,  c0, true);
        store_d_masked(msg, S, row_base, N, 16, c1, true);
    }
}

// agg[dst] += msg[src]; EPW edges per wave, lane = column; edge ids scalarized.
__global__ __launch_bounds__(256) void edge_kernel(const float* __restrict__ msg,
                                                   const long long* __restrict__ ei,
                                                   float* __restrict__ agg, long long E) {
    const int lane = threadIdx.x & 31;
    const int wv = __builtin_amdgcn_readfirstlane((int)(threadIdx.x >> 5));
    const long long e0 = ((long long)blockIdx.x * 8 + wv) * EPW;
    if (e0 >= E) return;
    // stream-ahead prefetch of the edge-index arrays (global_prefetch_b8)
    __builtin_prefetch((const void*)(ei + e0 + 4096), 0, 1);
    __builtin_prefetch((const void*)(ei + E + e0 + 4096), 0, 1);
#pragma unroll
    for (int k = 0; k < EPW; ++k) {
        long long e = e0 + k;
        if (e >= E) return;                    // wave-uniform
        int src = __builtin_amdgcn_readfirstlane((int)ei[e]);
        int dst = __builtin_amdgcn_readfirstlane((int)ei[E + e]);
        float v = msg[(size_t)src * S + lane];                 // coalesced 128B
        atomicAdd(agg + (size_t)dst * S + lane, v);            // coalesced 128B, L2
    }
}

// GRU cell + residual, in place on state.
__global__ __launch_bounds__(256) void gru_kernel(float* __restrict__ state,
                                                  const float* __restrict__ agg,
                                                  const _Float16* __restrict__ packed,
                                                  int fb,
                                                  const float* __restrict__ bih,
                                                  const float* __restrict__ bhh, int N) {
    __shared__ __align__(16) float lds[8][16 * LDS_PITCH];
    int wave = blockIdx.x * (blockDim.x >> 5) + (threadIdx.x >> 5);
    int row_base = wave * 16;
    if (row_base >= N) return;                 // wave-uniform
    const int lane = threadIdx.x & 31;
    const int jl   = lane & 15;
    const int mofs = (lane & 16) ? 8 : 0;
    const bool full = (row_base + 16 <= N);

    v16h a_i = load_a_f16(agg,   row_base, N);
    v16h a_h = load_a_f16(state, row_base, N);

    v8f out0, out1;
#pragma unroll
    for (int t = 0; t < 2; ++t) {
        int jb = t * 16;
        // frag(wih, gate g, tile t) = fb + g*2 + t ; whh at +6
        v8f gi_r = wmma16(a_i, load_packed(packed, fb + 0 + t),     splat8(bih[0 * S + jb + jl]));
        v8f gh_r = wmma16(a_h, load_packed(packed, fb + 6 + 0 + t), splat8(bhh[0 * S + jb + jl]));
        v8f gi_z = wmma16(a_i, load_packed(packed, fb + 2 + t),     splat8(bih[1 * S + jb + jl]));
        v8f gh_z = wmma16(a_h, load_packed(packed, fb + 6 + 2 + t), splat8(bhh[1 * S + jb + jl]));
        v8f gi_n = wmma16(a_i, load_packed(packed, fb + 4 + t),     splat8(bih[2 * S + jb + jl]));
        v8f gh_n = wmma16(a_h, load_packed(packed, fb + 6 + 4 + t), splat8(bhh[2 * S + jb + jl]));

        v8f o;
        if (full) {
            const float* pst = state + (size_t)(row_base + mofs) * S + jb + jl;
#pragma unroll
            for (int v = 0; v < 8; ++v) {
                float st = pst[(size_t)v * S];
                float r  = sigmoidf(gi_r[v] + gh_r[v]);
                float z  = sigmoidf(gi_z[v] + gh_z[v]);
                float nn = tanhf(gi_n[v] + r * gh_n[v]);
                o[v] = st + (1.0f - z) * nn + z * st;          // state + h_new
            }
        } else {
#pragma unroll
            for (int v = 0; v < 8; ++v) {
                int m  = row_base + mofs + v;
                int mm = (m < N) ? m : (N - 1);
                float st = state[(size_t)mm * S + jb + jl];
                float r  = sigmoidf(gi_r[v] + gh_r[v]);
                float z  = sigmoidf(gi_z[v] + gh_z[v]);
                float nn = tanhf(gi_n[v] + r * gh_n[v]);
                if (m < N) state[(size_t)m * S + jb + jl] = st + (1.0f - z) * nn + z * st;
            }
        }
        if (t == 0) out0 = o; else out1 = o;
    }
    if (full)
        store_tile32_coalesced(state, row_base, out0, out1, false, lds[threadIdx.x >> 5]);
}

// mu = state @ mu_W + mu_b ; logstd = state @ ls_W + ls_b
__global__ __launch_bounds__(256) void head_kernel(const float* __restrict__ state,
                                                   const _Float16* __restrict__ packed,
                                                   const float* __restrict__ mub,
                                                   const float* __restrict__ lsb,
                                                   float* __restrict__ out, int N) {
    int wave = blockIdx.x * (blockDim.x >> 5) + (threadIdx.x >> 5);
    int row_base = wave * 16;
    if (row_base >= N) return;                 // wave-uniform
    const int jl = threadIdx.x & 15;
    const bool full = (row_base + 16 <= N);
    v16h a = load_a_f16(state, row_base, N);

    v8f cm = wmma16(a, load_packed(packed, 42), splat8(mub[jl]));
    v8f cl = wmma16(a, load_packed(packed, 43), splat8(lsb[jl]));
    if (full) {
        store_d_full(out,                   16, row_base, 0, cm, false);
        store_d_full(out + (size_t)N * 16,  16, row_base, 0, cl, false);
    } else {
        store_d_masked(out,                  16, row_base, N, 0, cm, false);
        store_d_masked(out + (size_t)N * 16, 16, row_base, N, 0, cl, false);
    }
}

// ---------------- launcher ----------------

extern "C" void kernel_launch(void* const* d_in, const int* in_sizes, int n_in,
                              void* d_out, int out_size, void* d_ws, size_t ws_size,
                              hipStream_t stream) {
    const float*     x     = (const float*)d_in[0];
    const long long* ei    = (const long long*)d_in[1];   // int64 edge_index [2,E]
    const float*     W_in  = (const float*)d_in[2];
    const float*     b_in  = (const float*)d_in[3];
    const float*     msg_W = (const float*)d_in[4];       // [R,32,32]
    const float*     msg_b = (const float*)d_in[5];       // [R,32]
    const float*     g_wih = (const float*)d_in[6];       // [R,96,32]
    const float*     g_whh = (const float*)d_in[7];       // [R,96,32]
    const float*     g_bih = (const float*)d_in[8];       // [R,96]
    const float*     g_bhh = (const float*)d_in[9];       // [R,96]
    const float*     muW   = (const float*)d_in[10];
    const float*     mub   = (const float*)d_in[11];
    const float*     lsW   = (const float*)d_in[12];
    const float*     lsb   = (const float*)d_in[13];

    const int       N = in_sizes[0] / 7;
    const long long E = (long long)in_sizes[1] / 2;

    float* state = (float*)d_ws;                       // N*S floats
    float* msg   = state + (size_t)N * S;              // N*S floats
    float* agg   = msg   + (size_t)N * S;              // N*S floats
    _Float16* packed = (_Float16*)(agg + (size_t)N * S);  // NFRAG*32*16 halves

    const size_t ns   = (size_t)N * S;
    const int tiles   = (N + 15) / 16;
    const int gblocks = (tiles + 7) / 8;               // 8 waves/block, 1 tile/wave
    const unsigned eblocks = (unsigned)((E + 8 * EPW - 1) / (8 * EPW));

    pack_b_kernel<<<(NFRAG + 7) / 8, 256, 0, stream>>>(msg_W, g_wih, g_whh, muW, lsW, packed);
    input_net_kernel<<<(unsigned)((ns + 255) / 256), 256, 0, stream>>>(x, W_in, b_in, state, N);

    for (int r = 0; r < 3; ++r) {
        message_kernel<<<gblocks, 256, 0, stream>>>(state, packed, r * 2,
                                                    msg_b + r * S, msg, agg, N);
        edge_kernel<<<eblocks, 256, 0, stream>>>(msg, ei, agg, E);
        gru_kernel<<<gblocks, 256, 0, stream>>>(state, agg, packed, 6 + r * 12,
                                                g_bih + (size_t)r * 3 * S,
                                                g_bhh + (size_t)r * 3 * S, N);
    }

    head_kernel<<<gblocks, 256, 0, stream>>>(state, packed, mub, lsb, (float*)d_out, N);
}
